// VariableSelectionNetwork_21827023798803
// MI455X (gfx1250) — matrix operand order repaired and checked by reading, not compiled, not run on hardware
//
#include <hip/hip_runtime.h>

typedef __attribute__((ext_vector_type(16))) _Float16 v16h;
typedef __attribute__((ext_vector_type(8)))  _Float16 v8h;
typedef __attribute__((ext_vector_type(8)))  float    v8f;
typedef __attribute__((ext_vector_type(4)))  float    v4f;

#define B_    32
#define T_    256
#define V_    16
#define D_    64
#define H_    256
#define C_    128
#define F_    1024
#define NTOK  8192
#define MT    2                 /* M-tiles (of 16 rows) per workgroup        */
#define MR    (MT * 16)         /* rows per workgroup = 32                   */
#define LDH   264               /* padded f16 LDS row stride (halves)        */
#define LDF   260               /* padded f32 LDS row stride                 */
#define EPSF  1e-5f

// ---------------------------------------------------------------------------
// WMMA helpers (CDNA5 v_wmma_f32_16x16x32_f16, wave32)
// ---------------------------------------------------------------------------
__device__ __forceinline__ v8f wmma_f16(v16h a, v16h b, v8f c) {
  return __builtin_amdgcn_wmma_f32_16x16x32_f16(false, a, false, b, (short)0, c,
                                                false, false);
}

// A fragment (16x32 f16) from an f32 row in global memory.
__device__ __forceinline__ v16h a_frag_g32(const float* rowp, int kbase, int lane) {
  int ko = kbase + ((lane >> 4) << 3);
  v4f f0 = *(const v4f*)(rowp + ko);
  v4f f1 = *(const v4f*)(rowp + ko + 4);
  v4f f2 = *(const v4f*)(rowp + ko + 16);
  v4f f3 = *(const v4f*)(rowp + ko + 20);
  v16h a;
#pragma unroll
  for (int i = 0; i < 4; ++i) {
    a[i]      = (_Float16)f0[i];
    a[4 + i]  = (_Float16)f1[i];
    a[8 + i]  = (_Float16)f2[i];
    a[12 + i] = (_Float16)f3[i];
  }
  return a;
}

// A fragment from a row-major f16 LDS tile (stride LDH halves).
__device__ __forceinline__ v16h a_frag_lds(const _Float16* h, int kbase, int lane) {
  const _Float16* p = h + (lane & 15) * LDH + kbase + ((lane >> 4) << 3);
  v8h lo = *(const v8h*)p;
  v8h hi = *(const v8h*)(p + 16);
  v16h a;
#pragma unroll
  for (int i = 0; i < 8; ++i) { a[i] = lo[i]; a[8 + i] = hi[i]; }
  return a;
}

// B fragment (32x16 f16) from an f16 transposed weight matrix wT (N x K).
__device__ __forceinline__ v16h b_frag(const _Float16* wT, int Kdim, int n,
                                       int kbase, int lane) {
  const _Float16* p = wT + (size_t)(n + (lane & 15)) * Kdim + kbase +
                      ((lane >> 4) << 3);
  v8h lo = *(const v8h*)p;
  v8h hi = *(const v8h*)(p + 16);
  v16h b;
#pragma unroll
  for (int i = 0; i < 8; ++i) { b[i] = lo[i]; b[8 + i] = hi[i]; }
  return b;
}

__device__ __forceinline__ float eluf(float x) { return x > 0.f ? x : (__expf(x) - 1.f); }
__device__ __forceinline__ float sigm(float x) { return 1.f / (1.f + __expf(-x)); }

// ---------------------------------------------------------------------------
// Prologue: batched convert f32 (K x N) -> f16 transposed (N x K)
// ---------------------------------------------------------------------------
__global__ void cvtT_kernel(const float* __restrict__ src, _Float16* __restrict__ dst,
                            int K, int N, long long total) {
  long long i = (long long)blockIdx.x * blockDim.x + threadIdx.x;
  if (i >= total) return;
  long long mn = (long long)K * N;
  long long m = i / mn;
  long long r = i - m * mn;
  int n = (int)(r / K);
  int k = (int)(r - (long long)n * K);
  dst[m * mn + (long long)n * K + k] = (_Float16)src[m * mn + (long long)k * N + n];
}

__global__ void ctx_kernel(const float* __restrict__ context,
                           const float* __restrict__ ctx_w,
                           float* __restrict__ ctxp) {
  int b = blockIdx.x, h = threadIdx.x;
  float s = 0.f;
#pragma unroll 8
  for (int c = 0; c < C_; ++c) s += context[b * C_ + c] * ctx_w[c * H_ + h];
  ctxp[b * H_ + h] = s;
}

// ---------------------------------------------------------------------------
// Weight GRN: 32-token tile / workgroup, 8 waves.
// ---------------------------------------------------------------------------
__global__ __launch_bounds__(256, 2) void wgrn_kernel(
    const float* __restrict__ x,           // (NTOK, F)
    const _Float16* __restrict__ fc1T,     // (H, F)
    const float* __restrict__ fc1_b,
    const float* __restrict__ ctxp,        // (B, H)
    const _Float16* __restrict__ fc2T,     // (H, H)
    const float* __restrict__ fc2_b,
    const _Float16* __restrict__ gluT,     // (2V, H)
    const float* __restrict__ glu_b,
    const _Float16* __restrict__ skipT,    // (V, F)
    const float* __restrict__ skip_b,
    const float* __restrict__ ln_g,
    const float* __restrict__ ln_b,
    float* __restrict__ w_ws,              // (NTOK, V)
    float* __restrict__ w_out)             // (NTOK, V)
{
  __shared__ _Float16 h1[MR * LDH];
  __shared__ _Float16 h2[MR * LDH];
  __shared__ float res_part[8][MR][16];
  __shared__ float resbuf[MR][16];
  __shared__ float gbuf[MR][32];

  const int tid  = threadIdx.x;
  const int wave = tid >> 5;
  const int lane = tid & 31;
  const int tok0 = blockIdx.x * MR;
  const int bidx = tok0 >> 8;
  const int n0   = wave * 32;
  const int hi8  = (lane >> 4) << 3;

  const float* xrow0 = x + (size_t)(tok0 + (lane & 15)) * F_;
  const float* xrow1 = xrow0 + (size_t)16 * F_;

  // ---- fc1 (K=1024) + skip (K split across waves) ----
  v8f cf[MT][2];
  v8f cs[MT];
#pragma unroll
  for (int mt = 0; mt < MT; ++mt) { cf[mt][0] = {}; cf[mt][1] = {}; cs[mt] = {}; }
  for (int ks = 0; ks < 32; ++ks) {
    v16h a0 = a_frag_g32(xrow0, ks * 32, lane);
    v16h a1 = a_frag_g32(xrow1, ks * 32, lane);
#pragma unroll
    for (int nt = 0; nt < 2; ++nt) {
      v16h b = b_frag(fc1T, F_, n0 + nt * 16, ks * 32, lane);
      cf[0][nt] = wmma_f16(a0, b, cf[0][nt]);
      cf[1][nt] = wmma_f16(a1, b, cf[1][nt]);
    }
    if ((ks >> 2) == wave) {
      v16h bs = b_frag(skipT, F_, 0, ks * 32, lane);
      cs[0] = wmma_f16(a0, bs, cs[0]);
      cs[1] = wmma_f16(a1, bs, cs[1]);
    }
  }
#pragma unroll
  for (int mt = 0; mt < MT; ++mt)
#pragma unroll
    for (int j = 0; j < 8; ++j)
      res_part[wave][mt * 16 + j + hi8][lane & 15] = cs[mt][j];
#pragma unroll
  for (int mt = 0; mt < MT; ++mt)
#pragma unroll
    for (int nt = 0; nt < 2; ++nt)
#pragma unroll
      for (int j = 0; j < 8; ++j) {
        int row = mt * 16 + j + hi8, col = n0 + nt * 16 + (lane & 15);
        float v = cf[mt][nt][j] + fc1_b[col] + ctxp[bidx * H_ + col];
        h1[row * LDH + col] = (_Float16)eluf(v);
      }
  __syncthreads();

  // residual reduce: 512 entries / 256 threads
#pragma unroll
  for (int rep = 0; rep < 2; ++rep) {
    int r = (tid >> 4) + rep * 16, c = tid & 15;
    float s = skip_b[c];
#pragma unroll
    for (int w = 0; w < 8; ++w) s += res_part[w][r][c];
    resbuf[r][c] = s;
  }

  // ---- fc2 (K=256) ----
  v8f c2[MT][2];
#pragma unroll
  for (int mt = 0; mt < MT; ++mt) { c2[mt][0] = {}; c2[mt][1] = {}; }
#pragma unroll
  for (int ks = 0; ks < 8; ++ks) {
    v16h a0 = a_frag_lds(h1, ks * 32, lane);
    v16h a1 = a_frag_lds(h1 + 16 * LDH, ks * 32, lane);
#pragma unroll
    for (int nt = 0; nt < 2; ++nt) {
      v16h b = b_frag(fc2T, H_, n0 + nt * 16, ks * 32, lane);
      c2[0][nt] = wmma_f16(a0, b, c2[0][nt]);
      c2[1][nt] = wmma_f16(a1, b, c2[1][nt]);
    }
  }
#pragma unroll
  for (int mt = 0; mt < MT; ++mt)
#pragma unroll
    for (int nt = 0; nt < 2; ++nt)
#pragma unroll
      for (int j = 0; j < 8; ++j) {
        int row = mt * 16 + j + hi8, col = n0 + nt * 16 + (lane & 15);
        h2[row * LDH + col] = (_Float16)(c2[mt][nt][j] + fc2_b[col]);
      }
  __syncthreads();

  // ---- GLU projection (K=256, N=32): waves 0 and 1 ----
  if (wave < 2) {
    v8f g[MT]; g[0] = {}; g[1] = {};
    int nb = wave * 16;
#pragma unroll
    for (int ks = 0; ks < 8; ++ks) {
      v16h b = b_frag(gluT, H_, nb, ks * 32, lane);
      g[0] = wmma_f16(a_frag_lds(h2, ks * 32, lane), b, g[0]);
      g[1] = wmma_f16(a_frag_lds(h2 + 16 * LDH, ks * 32, lane), b, g[1]);
    }
#pragma unroll
    for (int mt = 0; mt < MT; ++mt)
#pragma unroll
      for (int j = 0; j < 8; ++j) {
        int row = mt * 16 + j + hi8, col = nb + (lane & 15);
        gbuf[row][col] = g[mt][j] + glu_b[col];
      }
  }
  __syncthreads();

  // ---- GLU + residual + LN + softmax, one row per thread ----
  if (tid < MR) {
    float hrow[16];
    float mu = 0.f;
#pragma unroll
    for (int c = 0; c < 16; ++c) {
      float a = gbuf[tid][c], bg = gbuf[tid][16 + c];
      hrow[c] = a * sigm(bg) + resbuf[tid][c];
      mu += hrow[c];
    }
    mu *= (1.f / 16.f);
    float var = 0.f;
#pragma unroll
    for (int c = 0; c < 16; ++c) { float d = hrow[c] - mu; var += d * d; }
    var *= (1.f / 16.f);
    float rs = rsqrtf(var + EPSF);
    float mx = -3.0e38f;
#pragma unroll
    for (int c = 0; c < 16; ++c) {
      hrow[c] = (hrow[c] - mu) * rs * ln_g[c] + ln_b[c];
      mx = fmaxf(mx, hrow[c]);
    }
    float den = 0.f;
#pragma unroll
    for (int c = 0; c < 16; ++c) { hrow[c] = __expf(hrow[c] - mx); den += hrow[c]; }
    float inv = 1.f / den;
    int tok = tok0 + tid;
#pragma unroll
    for (int c = 0; c < 16; ++c) {
      float w = hrow[c] * inv;
      w_ws[tok * V_ + c] = w;
      w_out[tok * V_ + c] = w;
    }
  }
}

// ---------------------------------------------------------------------------
// Per-variable GRNs + LN + weighted combine. 32-token tile / workgroup.
// ---------------------------------------------------------------------------
__global__ __launch_bounds__(256, 2) void vargrn_kernel(
    const float* __restrict__ x,           // (NTOK, V, D)
    const _Float16* __restrict__ fc1T,     // (V, H, D)
    const float* __restrict__ fc1_b,
    const _Float16* __restrict__ fc2T,     // (V, H, H)
    const float* __restrict__ fc2_b,
    const _Float16* __restrict__ gluT,     // (V, 2H, H)
    const float* __restrict__ glu_b,
    const _Float16* __restrict__ skipT,    // (V, H, D)
    const float* __restrict__ skip_b,
    const float* __restrict__ ln_g,
    const float* __restrict__ ln_b,
    const float* __restrict__ weights,     // (NTOK, V)
    float* __restrict__ selected)          // (NTOK, H)
{
  __shared__ _Float16 h1[MR * LDH];
  __shared__ _Float16 h2[MR * LDH];
  __shared__ float hb[MR * LDF];
  __shared__ float red[2][MR][8];
  __shared__ float stats[2][MR];
  __shared__ float wtile[MR][16];

  const int tid  = threadIdx.x;
  const int wave = tid >> 5;
  const int lane = tid & 31;
  const int tok0 = blockIdx.x * MR;
  const int n0   = wave * 32;
  const int hi8  = (lane >> 4) << 3;
  const int rr   = tid >> 3;     // 0..31: row in element-wise phases
  const int seg  = tid & 7;      // 0..7: 32-col segment owned

  float accr[32];
#pragma unroll
  for (int i = 0; i < 32; ++i) accr[i] = 0.f;

#pragma unroll
  for (int rep = 0; rep < 2; ++rep) {
    int r = (tid >> 4) + rep * 16;
    wtile[r][tid & 15] = weights[(tok0 + r) * V_ + (tid & 15)];
  }
  __syncthreads();

  for (int v = 0; v < V_; ++v) {
    const _Float16* fc1T_v  = fc1T  + (size_t)v * H_ * D_;
    const _Float16* skipT_v = skipT + (size_t)v * H_ * D_;
    const _Float16* fc2T_v  = fc2T  + (size_t)v * H_ * H_;
    const _Float16* gluT_v  = gluT  + (size_t)v * 2 * H_ * H_;
    __builtin_prefetch(fc2T_v + (size_t)tid * 64, 0, 1);
    __builtin_prefetch(gluT_v + (size_t)tid * 64, 0, 1);

    // ---- fc1 + skip (K=64); each B fragment feeds both M-tiles ----
    v16h a[MT][2];
#pragma unroll
    for (int mt = 0; mt < MT; ++mt) {
      const float* xrow =
          x + ((size_t)(tok0 + mt * 16 + (lane & 15)) * V_ + v) * D_;
      a[mt][0] = a_frag_g32(xrow, 0, lane);
      a[mt][1] = a_frag_g32(xrow, 32, lane);
    }
    v8f cf[MT][2], cs[MT][2];
#pragma unroll
    for (int mt = 0; mt < MT; ++mt) {
      cf[mt][0] = {}; cf[mt][1] = {}; cs[mt][0] = {}; cs[mt][1] = {};
    }
#pragma unroll
    for (int nt = 0; nt < 2; ++nt)
#pragma unroll
      for (int ks = 0; ks < 2; ++ks) {
        v16h bf = b_frag(fc1T_v, D_, n0 + nt * 16, ks * 32, lane);
        v16h bs = b_frag(skipT_v, D_, n0 + nt * 16, ks * 32, lane);
#pragma unroll
        for (int mt = 0; mt < MT; ++mt) {
          cf[mt][nt] = wmma_f16(a[mt][ks], bf, cf[mt][nt]);
          cs[mt][nt] = wmma_f16(a[mt][ks], bs, cs[mt][nt]);
        }
      }
    // h1 <- elu(fc1 + b);  hb <- skip + skip_b (parked in LDS, frees VGPRs)
#pragma unroll
    for (int mt = 0; mt < MT; ++mt)
#pragma unroll
      for (int nt = 0; nt < 2; ++nt)
#pragma unroll
        for (int j = 0; j < 8; ++j) {
          int row = mt * 16 + j + hi8, col = n0 + nt * 16 + (lane & 15);
          h1[row * LDH + col] = (_Float16)eluf(cf[mt][nt][j] + fc1_b[v * H_ + col]);
          hb[row * LDF + col] = cs[mt][nt][j] + skip_b[v * H_ + col];
        }
    __syncthreads();

    // ---- fc2 (K=256) ----
    v8f c2[MT][2];
#pragma unroll
    for (int mt = 0; mt < MT; ++mt) { c2[mt][0] = {}; c2[mt][1] = {}; }
#pragma unroll
    for (int ks = 0; ks < 8; ++ks) {
      v16h a0 = a_frag_lds(h1, ks * 32, lane);
      v16h a1 = a_frag_lds(h1 + 16 * LDH, ks * 32, lane);
#pragma unroll
      for (int nt = 0; nt < 2; ++nt) {
        v16h b = b_frag(fc2T_v, H_, n0 + nt * 16, ks * 32, lane);
        c2[0][nt] = wmma_f16(a0, b, c2[0][nt]);
        c2[1][nt] = wmma_f16(a1, b, c2[1][nt]);
      }
    }
#pragma unroll
    for (int mt = 0; mt < MT; ++mt)
#pragma unroll
      for (int nt = 0; nt < 2; ++nt)
#pragma unroll
        for (int j = 0; j < 8; ++j) {
          int row = mt * 16 + j + hi8, col = n0 + nt * 16 + (lane & 15);
          h2[row * LDH + col] = (_Float16)(c2[mt][nt][j] + fc2_b[v * H_ + col]);
        }
    __syncthreads();

    // ---- GLU (K=256, 4 N-tiles x 2 M-tiles) ----
    v8f cg[4][MT];
#pragma unroll
    for (int nt = 0; nt < 4; ++nt) { cg[nt][0] = {}; cg[nt][1] = {}; }
#pragma unroll
    for (int ks = 0; ks < 8; ++ks) {
      v16h a0 = a_frag_lds(h2, ks * 32, lane);
      v16h a1 = a_frag_lds(h2 + 16 * LDH, ks * 32, lane);
#pragma unroll
      for (int nt = 0; nt < 4; ++nt) {
        int ncol = n0 + (nt & 1) * 16 + ((nt >> 1) ? H_ : 0);
        v16h b = b_frag(gluT_v, H_, ncol, ks * 32, lane);
        cg[nt][0] = wmma_f16(a0, b, cg[nt][0]);
        cg[nt][1] = wmma_f16(a1, b, cg[nt][1]);
      }
    }
#pragma unroll
    for (int mt = 0; mt < MT; ++mt)
#pragma unroll
      for (int t = 0; t < 2; ++t)
#pragma unroll
        for (int j = 0; j < 8; ++j) {
          int row = mt * 16 + j + hi8, col = n0 + t * 16 + (lane & 15);
          float av = cg[t][mt][j]     + glu_b[v * 2 * H_ + col];
          float gv = cg[2 + t][mt][j] + glu_b[v * 2 * H_ + H_ + col];
          hb[row * LDF + col] = av * sigm(gv) + hb[row * LDF + col];
        }
    __syncthreads();

    // ---- LayerNorm over H=256 + weighted accumulate ----
    float s = 0.f, s2 = 0.f;
    int base = rr * LDF + seg * 32;
#pragma unroll
    for (int i = 0; i < 32; ++i) { float u = hb[base + i]; s += u; s2 += u * u; }
    red[0][rr][seg] = s;
    red[1][rr][seg] = s2;
    __syncthreads();
    if (tid < MR) {
      float ts = 0.f, ts2 = 0.f;
#pragma unroll
      for (int g = 0; g < 8; ++g) { ts += red[0][tid][g]; ts2 += red[1][tid][g]; }
      float mu = ts * (1.f / (float)H_);
      float var = ts2 * (1.f / (float)H_) - mu * mu;
      stats[0][tid] = mu;
      stats[1][tid] = rsqrtf(var + EPSF);
    }
    __syncthreads();
    {
      float mu = stats[0][rr], rs = stats[1][rr];
      float wv = wtile[rr][v];
#pragma unroll
      for (int i = 0; i < 32; ++i) {
        int c = seg * 32 + i;
        float u = hb[rr * LDF + c];
        accr[i] += wv * ((u - mu) * rs * ln_g[v * H_ + c] + ln_b[v * H_ + c]);
      }
    }
    __syncthreads();
  }

#pragma unroll
  for (int i = 0; i < 32; ++i)
    selected[(size_t)(tok0 + rr) * H_ + seg * 32 + i] = accr[i];
}

// ---------------------------------------------------------------------------
// Host launcher
// ---------------------------------------------------------------------------
extern "C" void kernel_launch(void* const* d_in, const int* in_sizes, int n_in,
                              void* d_out, int out_size, void* d_ws, size_t ws_size,
                              hipStream_t stream) {
  (void)in_sizes; (void)n_in; (void)out_size; (void)ws_size;
  const float* x         = (const float*)d_in[0];
  const float* context   = (const float*)d_in[1];
  const float* vg_fc1_w  = (const float*)d_in[2];
  const float* vg_fc1_b  = (const float*)d_in[3];
  const float* vg_fc2_w  = (const float*)d_in[4];
  const float* vg_fc2_b  = (const float*)d_in[5];
  const float* vg_glu_w  = (const float*)d_in[6];
  const float* vg_glu_b  = (const float*)d_in[7];
  const float* vg_skip_w = (const float*)d_in[8];
  const float* vg_skip_b = (const float*)d_in[9];
  const float* vg_ln_g   = (const float*)d_in[10];
  const float* vg_ln_b   = (const float*)d_in[11];
  const float* wg_fc1_w  = (const float*)d_in[12];
  const float* wg_fc1_b  = (const float*)d_in[13];
  const float* wg_ctx_w  = (const float*)d_in[14];
  const float* wg_fc2_w  = (const float*)d_in[15];
  const float* wg_fc2_b  = (const float*)d_in[16];
  const float* wg_glu_w  = (const float*)d_in[17];
  const float* wg_glu_b  = (const float*)d_in[18];
  const float* wg_skip_w = (const float*)d_in[19];
  const float* wg_skip_b = (const float*)d_in[20];
  const float* wg_ln_g   = (const float*)d_in[21];
  const float* wg_ln_b   = (const float*)d_in[22];

  float* out_selected = (float*)d_out;                       // (NTOK, H)
  float* out_weights  = (float*)d_out + (size_t)NTOK * H_;   // (NTOK, V)

  char* ws = (char*)d_ws;
  size_t off = 0;
  float* ctxp = (float*)(ws + off);        off += (size_t)B_ * H_ * 4;
  float* wbuf = (float*)(ws + off);        off += (size_t)NTOK * V_ * 4;
  _Float16* wg_fc1T  = (_Float16*)(ws + off); off += (size_t)F_ * H_ * 2;
  _Float16* wg_skipT = (_Float16*)(ws + off); off += (size_t)F_ * V_ * 2;
  _Float16* wg_fc2T  = (_Float16*)(ws + off); off += (size_t)H_ * H_ * 2;
  _Float16* wg_gluT  = (_Float16*)(ws + off); off += (size_t)H_ * 2 * V_ * 2;
  _Float16* vg_fc1T  = (_Float16*)(ws + off); off += (size_t)V_ * D_ * H_ * 2;
  _Float16* vg_skipT = (_Float16*)(ws + off); off += (size_t)V_ * D_ * H_ * 2;
  _Float16* vg_fc2T  = (_Float16*)(ws + off); off += (size_t)V_ * H_ * H_ * 2;
  _Float16* vg_gluT  = (_Float16*)(ws + off); off += (size_t)V_ * H_ * 2 * H_ * 2;

  auto launchT = [&](const float* src, _Float16* dst, int K, int N, int batch) {
    long long total = (long long)batch * K * N;
    int grid = (int)((total + 255) / 256);
    cvtT_kernel<<<grid, 256, 0, stream>>>(src, dst, K, N, total);
  };
  launchT(wg_fc1_w,  wg_fc1T,  F_, H_,     1);
  launchT(wg_skip_w, wg_skipT, F_, V_,     1);
  launchT(wg_fc2_w,  wg_fc2T,  H_, H_,     1);
  launchT(wg_glu_w,  wg_gluT,  H_, 2 * V_, 1);
  launchT(vg_fc1_w,  vg_fc1T,  D_, H_,     V_);
  launchT(vg_skip_w, vg_skipT, D_, H_,     V_);
  launchT(vg_fc2_w,  vg_fc2T,  H_, H_,     V_);
  launchT(vg_glu_w,  vg_gluT,  H_, 2 * H_, V_);

  ctx_kernel<<<B_, H_, 0, stream>>>(context, wg_ctx_w, ctxp);

  wgrn_kernel<<<NTOK / MR, 256, 0, stream>>>(
      x, wg_fc1T, wg_fc1_b, ctxp, wg_fc2T, wg_fc2_b, wg_gluT, wg_glu_b,
      wg_skipT, wg_skip_b, wg_ln_g, wg_ln_b, wbuf, out_weights);

  vargrn_kernel<<<NTOK / MR, 256, 0, stream>>>(
      x, vg_fc1T, vg_fc1_b, vg_fc2T, vg_fc2_b, vg_gluT, vg_glu_b,
      vg_skipT, vg_skip_b, vg_ln_g, vg_ln_b, wbuf, out_selected);
}
